// IOI_70506183131666
// MI455X (gfx1250) — matrix operand order repaired and checked by reading, not compiled, not run on hardware
//
#include <hip/hip_runtime.h>
#include <hip/hip_bf16.h>
#include <math.h>

// ---------------------------------------------------------------------------
// IOI model on gfx1250.  All GEMMs -> v_wmma_f32_16x16x32_f16 with:
//   * zero-padded K dims in workspace (no guards on A-side loads)
//   * weights pre-packed per launch to f16, transposed+padded (2x v8h loads)
//   * wide unconditional loads (float4 / v8h), value-select instead of exec
// ---------------------------------------------------------------------------

typedef __attribute__((ext_vector_type(16))) _Float16 v16h;
typedef __attribute__((ext_vector_type(8)))  _Float16 v8h;
typedef __attribute__((ext_vector_type(8)))  float    v8f;

#define DEV static __device__ __forceinline__

#define CP 224   // padded channel stride (C=200 -> 7 K-steps of 32)
#define SP 64    // padded attention-score row stride (L=50)

DEV v8f wmma16(v16h a, v16h b, v8f c) {
  return __builtin_amdgcn_wmma_f32_16x16x32_f16(false, a, false, b, (short)0, c, false, false);
}

DEV int imin(int a, int b) { return a < b ? a : b; }

DEV float wave_sum(float v) {
  for (int o = 16; o > 0; o >>= 1) v += __shfl_xor(v, o, 32);
  return v;
}
DEV float wave_max(float v) {
  for (int o = 16; o > 0; o >>= 1) v = fmaxf(v, __shfl_xor(v, o, 32));
  return v;
}

// A fragment: 16 f16 per lane from row base p (= row + kb + 16*half), f32 src.
// Layout: element s -> K = 16*half + s (contiguous, matches B fragments).
DEV v16h load_a_frag(const float* p) {
  float4 a0 = *(const float4*)(p);
  float4 a1 = *(const float4*)(p + 4);
  float4 a2 = *(const float4*)(p + 8);
  float4 a3 = *(const float4*)(p + 12);
  v16h av;
  av[0]  = (_Float16)a0.x; av[1]  = (_Float16)a0.y; av[2]  = (_Float16)a0.z; av[3]  = (_Float16)a0.w;
  av[4]  = (_Float16)a1.x; av[5]  = (_Float16)a1.y; av[6]  = (_Float16)a1.z; av[7]  = (_Float16)a1.w;
  av[8]  = (_Float16)a2.x; av[9]  = (_Float16)a2.y; av[10] = (_Float16)a2.z; av[11] = (_Float16)a2.w;
  av[12] = (_Float16)a3.x; av[13] = (_Float16)a3.y; av[14] = (_Float16)a3.z; av[15] = (_Float16)a3.w;
  return av;
}

// B fragment from pre-packed f16 Wt row (16 contiguous f16 at kb + 16*half).
DEV v16h load_b_frag16(const _Float16* p) {
  v8h w0 = *(const v8h*)(p);
  v8h w1 = *(const v8h*)(p + 8);
  v16h bv;
#pragma unroll
  for (int e = 0; e < 8; ++e) { bv[e] = w0[e]; bv[8 + e] = w1[e]; }
  return bv;
}

// B fragment from f32 activation row (A*B^T pattern), scaled by padded dvec.
DEV v16h load_b_frag_scaled(const float* p, const float* d) {
  float4 b0 = *(const float4*)(p);      float4 d0 = *(const float4*)(d);
  float4 b1 = *(const float4*)(p + 4);  float4 d1 = *(const float4*)(d + 4);
  float4 b2 = *(const float4*)(p + 8);  float4 d2 = *(const float4*)(d + 8);
  float4 b3 = *(const float4*)(p + 12); float4 d3 = *(const float4*)(d + 12);
  v16h bv;
  bv[0]  = (_Float16)(b0.x * d0.x); bv[1]  = (_Float16)(b0.y * d0.y);
  bv[2]  = (_Float16)(b0.z * d0.z); bv[3]  = (_Float16)(b0.w * d0.w);
  bv[4]  = (_Float16)(b1.x * d1.x); bv[5]  = (_Float16)(b1.y * d1.y);
  bv[6]  = (_Float16)(b1.z * d1.z); bv[7]  = (_Float16)(b1.w * d1.w);
  bv[8]  = (_Float16)(b2.x * d2.x); bv[9]  = (_Float16)(b2.y * d2.y);
  bv[10] = (_Float16)(b2.z * d2.z); bv[11] = (_Float16)(b2.w * d2.w);
  bv[12] = (_Float16)(b3.x * d3.x); bv[13] = (_Float16)(b3.y * d3.y);
  bv[14] = (_Float16)(b3.z * d3.z); bv[15] = (_Float16)(b3.w * d3.w);
  return bv;
}

// ---------------------------------------------------------------------------
// Weight packing:  Wt[n, k] = f16(W[k, n]), zero-padded to [NP, KP].
// ---------------------------------------------------------------------------
__global__ void pack_wt_kernel(const float* __restrict__ W, _Float16* __restrict__ Wt,
                               int Kd, int Nd, int KP, int NP) {
  int idx = blockIdx.x * blockDim.x + threadIdx.x;
  if (idx >= NP * KP) return;
  int n = idx / KP, k = idx % KP;
  float v = (n < Nd && k < Kd) ? W[(size_t)k * Nd + n] : 0.f;
  Wt[idx] = (_Float16)v;
}

// dvec padding: rows 0..11 = attn_d (flattened [4,3]), row 12 = ones.
__global__ void pack_dvec_kernel(const float* __restrict__ d, float* __restrict__ out) {
  int idx = blockIdx.x * blockDim.x + threadIdx.x;
  if (idx >= 13 * CP) return;
  int g = idx / CP, c = idx % CP;
  float v = 0.f;
  if (c < 200) v = (g < 12) ? d[g * 200 + c] : 1.f;
  out[idx] = v;
}

__global__ void zero_ws_kernel(float* __restrict__ p, size_t n) {
  size_t i = (size_t)blockIdx.x * blockDim.x + threadIdx.x;
  size_t stride = (size_t)gridDim.x * blockDim.x;
  for (; i < n; i += stride) p[i] = 0.f;
}

// ---------------------------------------------------------------------------
// GEMM:  Y[M,Nd] = act( X[M,*] @ Wt^T + bias ),  X f32 (ld = padded, zero pad),
// Wt f16 packed [NP,KP].  M multiple of 16; 4 waves/block, 1 tile/wave.
// ---------------------------------------------------------------------------
__global__ void gemm16_kernel(const float* __restrict__ X, int ldx,
                              const _Float16* __restrict__ Wt, int KP,
                              const float* __restrict__ bias,
                              float* __restrict__ Y, int ldy,
                              int M, int Nd, int act, int nt) {
  int lane = threadIdx.x & 31;
  int wave = threadIdx.x >> 5;
  int tn = blockIdx.x % nt;
  int tm = (blockIdx.x / nt) * 4 + wave;
  int r = lane & 15, half = lane >> 4;
  int row = tm * 16 + r;
  int col = tn * 16 + r;
  const float*    pa = X  + (size_t)row * ldx + 16 * half;
  const _Float16* pw = Wt + (size_t)col * KP + 16 * half;
  v8f acc = {};
  for (int kb = 0; kb < KP; kb += 32) {
    v16h av = load_a_frag(pa + kb);
    v16h bv = load_b_frag16(pw + kb);
    acc = wmma16(av, bv, acc);
  }
  int n = tn * 16 + (lane & 15);
  if (n < Nd) {
    float bb = bias ? bias[n] : 0.f;
#pragma unroll
    for (int q = 0; q < 8; ++q) {
      int m = tm * 16 + q + 8 * half;
      float v = acc[q] + bb;
      if (act) v = fmaxf(v, 0.f);
      Y[(size_t)m * ldy + n] = v;
    }
  }
}

// ---------------------------------------------------------------------------
// Concat GEMM:  A = [P0 | P1 | P2 | Pa*Pb] (virtual K=800, segments of 200),
// Y[M,200] = relu(A @ Wt^T + bias).  Segment select is branch-free (chunks of
// 8 never straddle a 200-boundary).
// ---------------------------------------------------------------------------
DEV void cat_chunk(const float* P0, const float* P1, const float* P2,
                   const float* Pa, const float* Pb, size_t rowoff, int c0,
                   float* out) {
  int s = (c0 >= 200) + (c0 >= 400) + (c0 >= 600);
  int lc = c0 - s * 200;
  const float* base = (s == 0) ? P0 : (s == 1) ? P1 : (s == 2) ? P2 : Pa;
  float4 x0 = *(const float4*)(base + rowoff + lc);
  float4 x1 = *(const float4*)(base + rowoff + lc + 4);
  float4 y0 = *(const float4*)(Pb + rowoff + lc);
  float4 y1 = *(const float4*)(Pb + rowoff + lc + 4);
  bool m3 = (s == 3);
  out[0] = x0.x * (m3 ? y0.x : 1.f); out[1] = x0.y * (m3 ? y0.y : 1.f);
  out[2] = x0.z * (m3 ? y0.z : 1.f); out[3] = x0.w * (m3 ? y0.w : 1.f);
  out[4] = x1.x * (m3 ? y1.x : 1.f); out[5] = x1.y * (m3 ? y1.y : 1.f);
  out[6] = x1.z * (m3 ? y1.z : 1.f); out[7] = x1.w * (m3 ? y1.w : 1.f);
}

__global__ void gemm_cat4_kernel(const float* __restrict__ P0, const float* __restrict__ P1,
                                 const float* __restrict__ P2, const float* __restrict__ Pa,
                                 const float* __restrict__ Pb,
                                 const _Float16* __restrict__ Wt,  // [208, 800]
                                 const float* __restrict__ bias,
                                 float* __restrict__ Y, int nt) {
  int lane = threadIdx.x & 31;
  int wave = threadIdx.x >> 5;
  int tn = blockIdx.x % nt;
  int tm = (blockIdx.x / nt) * 4 + wave;
  int r = lane & 15, half = lane >> 4;
  int row = tm * 16 + r;
  int col = tn * 16 + r;
  size_t rowoff = (size_t)row * CP;
  const _Float16* pw = Wt + (size_t)col * 800 + 16 * half;
  v8f acc = {};
  for (int kb = 0; kb < 800; kb += 32) {
    float lo[8], hi[8];
    cat_chunk(P0, P1, P2, Pa, Pb, rowoff, kb + 16 * half, lo);
    cat_chunk(P0, P1, P2, Pa, Pb, rowoff, kb + 16 * half + 8, hi);
    v16h av;
#pragma unroll
    for (int e = 0; e < 8; ++e) {
      av[e] = (_Float16)lo[e];
      av[8 + e] = (_Float16)hi[e];
    }
    v16h bv = load_b_frag16(pw + kb);
    acc = wmma16(av, bv, acc);
  }
  int n = tn * 16 + (lane & 15);
  if (n < 200) {
    float bb = bias[n];
#pragma unroll
    for (int q = 0; q < 8; ++q) {
      int m = tm * 16 + q + 8 * half;
      Y[(size_t)m * CP + n] = fmaxf(acc[q] + bb, 0.f);
    }
  }
}

// ---------------------------------------------------------------------------
// Batched bilinear scores:
//   out[b,i,j] = scale * sum_c A[b,i,c] * dpad[c] * B[b,j,c]
// A,B: [640, 50, CP] zero-padded.  out row stride/elem stride parametrized.
// ---------------------------------------------------------------------------
__global__ void attn_score_kernel(const float* __restrict__ A, const float* __restrict__ B,
                                  const float* __restrict__ dpad, float scale,
                                  float* __restrict__ out, int rowstride,
                                  int estride, int eoff) {
  int lane = threadIdx.x & 31;
  int b = blockIdx.x >> 4;
  int t = blockIdx.x & 15;
  int tm = t >> 2, tn = t & 3;
  int r = lane & 15, half = lane >> 4;
  int ic = imin(tm * 16 + r, 49);
  int jc = imin(tn * 16 + r, 49);
  const float* pa = A + ((size_t)b * 50 + ic) * CP + 16 * half;
  const float* pb = B + ((size_t)b * 50 + jc) * CP + 16 * half;
  const float* pd = dpad + 16 * half;
  v8f acc = {};
  for (int kb = 0; kb < CP; kb += 32) {
    v16h av = load_a_frag(pa + kb);
    v16h bv = load_b_frag_scaled(pb + kb, pd + kb);
    acc = wmma16(av, bv, acc);
  }
  int n = tn * 16 + (lane & 15);
  if (n < 50) {
#pragma unroll
    for (int q = 0; q < 8; ++q) {
      int m = tm * 16 + q + 8 * half;
      if (m < 50)
        out[((size_t)(b * 50 + m) * rowstride + n) * estride + eoff] = acc[q] * scale;
    }
  }
}

// ---------------------------------------------------------------------------
// Batched A(50x50, padded SP) @ V(50xC, padded CP): out[b,i,c], padded CP.
// ---------------------------------------------------------------------------
__global__ void attn_av_kernel(const float* __restrict__ S, const float* __restrict__ V,
                               float* __restrict__ out) {
  int lane = threadIdx.x & 31;
  int b = blockIdx.x / 52;
  int t = blockIdx.x % 52;
  int tm = t / 13, tn = t % 13;
  int r = lane & 15, half = lane >> 4;
  int ic = imin(tm * 16 + r, 49);
  int nc = imin(tn * 16 + r, 199);
  const float* pa = S + ((size_t)b * 50 + ic) * SP + 16 * half;
  const float* Vb = V + (size_t)b * 50 * CP;
  v8f acc = {};
#pragma unroll
  for (int kb = 0; kb < 64; kb += 32) {
    v16h av = load_a_frag(pa + kb);   // S pad cols [50,64) are zero
    v16h bv;
#pragma unroll
    for (int p = 0; p < 8; ++p) {
      int kv = kb + 16 * half + 2 * p;
      float b0 = Vb[(size_t)imin(kv, 49) * CP + nc];
      float b1 = Vb[(size_t)imin(kv + 1, 49) * CP + nc];
      b0 = (kv < 50) ? b0 : 0.f;
      b1 = (kv + 1 < 50) ? b1 : 0.f;
      bv[2 * p] = (_Float16)b0;
      bv[2 * p + 1] = (_Float16)b1;
    }
    acc = wmma16(av, bv, acc);
  }
  int nn = tn * 16 + (lane & 15);
  if (nn < 200) {
#pragma unroll
    for (int q = 0; q < 8; ++q) {
      int m = tm * 16 + q + 8 * half;
      if (m < 50) out[((size_t)b * 50 + m) * CP + nn] = acc[q];
    }
  }
}

// ---------------------------------------------------------------------------
// Masked softmax over S rows (stride SP); one wave per row.
// ---------------------------------------------------------------------------
__global__ void softmax_mask_kernel(float* __restrict__ S, const float* __restrict__ qm,
                                    const float* __restrict__ km) {
  int row = blockIdx.x;
  int b = row / 50;
  int lane = threadIdx.x;
  float* Sr = S + (size_t)row * SP;
  const float* kmr = km + (size_t)b * 50;
  const float NEGV = -4294967295.0f;  // -2^32 + 1
  float v0 = -3.0e38f, v1 = -3.0e38f;
  int j0 = lane, j1 = lane + 32;
  if (j0 < 50) v0 = (kmr[j0] == 0.f) ? NEGV : Sr[j0];
  if (j1 < 50) v1 = (kmr[j1] == 0.f) ? NEGV : Sr[j1];
  float mx = wave_max(fmaxf(v0, v1));
  float e0 = (j0 < 50) ? __expf(v0 - mx) : 0.f;
  float e1 = (j1 < 50) ? __expf(v1 - mx) : 0.f;
  float s = wave_sum(e0 + e1);
  float inv = qm[row] / s;
  if (j0 < 50) Sr[j0] = e0 * inv;
  if (j1 < 50) Sr[j1] = e1 * inv;
}

// ---------------------------------------------------------------------------
// out[row] = LN(A[row] + Br[row]) * g + beta   (C=200, stride CP, eps 1e-6)
// ---------------------------------------------------------------------------
__global__ void ln_res_kernel(float* __restrict__ out, const float* __restrict__ A,
                              const float* __restrict__ Br, const float* __restrict__ g,
                              const float* __restrict__ beta) {
  int row = blockIdx.x;
  int lane = threadIdx.x;
  size_t off = (size_t)row * CP;
  float x[7];
  float s = 0.f;
#pragma unroll
  for (int j = 0; j < 7; ++j) {
    int c = lane + 32 * j;
    x[j] = (c < 200) ? (A[off + c] + Br[off + c]) : 0.f;
    s += x[j];
  }
  float mean = wave_sum(s) * (1.f / 200.f);
  float v = 0.f;
#pragma unroll
  for (int j = 0; j < 7; ++j) {
    int c = lane + 32 * j;
    if (c < 200) { float d = x[j] - mean; v += d * d; }
  }
  float rs = rsqrtf(wave_sum(v) * (1.f / 200.f) + 1e-6f);
#pragma unroll
  for (int j = 0; j < 7; ++j) {
    int c = lane + 32 * j;
    if (c < 200) out[off + c] = (x[j] - mean) * rs * g[c] + beta[c];
  }
}

__global__ void add_inplace_kernel(float* __restrict__ dst, const float* __restrict__ src,
                                   int n) {
  int i = blockIdx.x * blockDim.x + threadIdx.x;
  if (i < n) dst[i] += src[i];
}

__global__ void zero_kernel(float* __restrict__ p, int n) {
  int i = blockIdx.x * blockDim.x + threadIdx.x;
  if (i < n) p[i] = 0.f;
}

// ---------------------------------------------------------------------------
// Embedding gather + masking.  ctx/resp rows stride CP (pad already zero).
// ---------------------------------------------------------------------------
__global__ void embed_kernel(const int* __restrict__ hist, const int* __restrict__ resp,
                             const float* __restrict__ emb,
                             float* __restrict__ ctx, float* __restrict__ rsp,
                             float* __restrict__ cm, float* __restrict__ rm) {
  int idx = blockIdx.x;  // n*50 + l,  n = b*10 + t
  int l = idx % 50;
  int n = idx / 50;
  int b = n / 10;
  int hid = hist[idx];
  int rid = resp[b * 50 + l];
  int c = threadIdx.x;
  if (c < 200) {
    ctx[(size_t)idx * CP + c] = hid ? emb[(size_t)hid * 200 + c] : 0.f;
    rsp[(size_t)idx * CP + c] = rid ? emb[(size_t)rid * 200 + c] : 0.f;
  }
  if (c == 0) {
    cm[idx] = hid ? 1.f : 0.f;
    rm[idx] = rid ? 1.f : 0.f;
  }
}

// ---------------------------------------------------------------------------
// Fused conv3x3(SAME,relu) + maxpool3x3/3(SAME).
// ---------------------------------------------------------------------------
__global__ void conv1_pool_kernel(const float* __restrict__ feat, const float* __restrict__ w,
                                  const float* __restrict__ bias, float* __restrict__ out) {
  int idx = blockIdx.x * blockDim.x + threadIdx.x;
  const int total = 640 * 17 * 17 * 32;
  if (idx >= total) return;
  int f = idx & 31;
  int t = idx >> 5;
  int px = t % 17; t /= 17;
  int py = t % 17;
  int n = t / 17;
  int y1 = 3 * py + 3; if (y1 > 50) y1 = 50;
  int x1 = 3 * px + 3; if (x1 > 50) x1 = 50;
  float best = -3.0e38f;
  for (int y = 3 * py; y < y1; ++y) {
    for (int x = 3 * px; x < x1; ++x) {
      float acc = bias[f];
      for (int dy = 0; dy < 3; ++dy) {
        int yy = y + dy - 1; if (yy < 0 || yy >= 50) continue;
        for (int dx = 0; dx < 3; ++dx) {
          int xx = x + dx - 1; if (xx < 0 || xx >= 50) continue;
          const float* fp = feat + ((size_t)(n * 50 + yy) * 50 + xx) * 3;
          const float* wp = w + ((dy * 3 + dx) * 3) * 32 + f;
          acc += fp[0] * wp[0] + fp[1] * wp[32] + fp[2] * wp[64];
        }
      }
      best = fmaxf(best, fmaxf(acc, 0.f));
    }
  }
  out[idx] = best;
}

__global__ void conv2_pool_kernel(const float* __restrict__ in, const float* __restrict__ w,
                                  const float* __restrict__ bias, float* __restrict__ out) {
  int idx = blockIdx.x * blockDim.x + threadIdx.x;
  const int total = 640 * 6 * 6 * 16;
  if (idx >= total) return;
  int f = idx & 15;
  int t = idx >> 4;
  int px = t % 6; t /= 6;
  int py = t % 6;
  int n = t / 6;
  int y1 = 3 * py + 3; if (y1 > 17) y1 = 17;
  int x1 = 3 * px + 3; if (x1 > 17) x1 = 17;
  float best = -3.0e38f;
  for (int y = 3 * py; y < y1; ++y) {
    for (int x = 3 * px; x < x1; ++x) {
      float acc = bias[f];
      for (int dy = 0; dy < 3; ++dy) {
        int yy = y + dy - 1; if (yy < 0 || yy >= 17) continue;
        for (int dx = 0; dx < 3; ++dx) {
          int xx = x + dx - 1; if (xx < 0 || xx >= 17) continue;
          const float* ip = in + ((size_t)(n * 17 + yy) * 17 + xx) * 32;
          const float* wp = w + ((dy * 3 + dx) * 32) * 16 + f;
#pragma unroll
          for (int i = 0; i < 32; ++i) acc += ip[i] * wp[i * 16];
        }
      }
      best = fmaxf(best, fmaxf(acc, 0.f));
    }
  }
  out[idx] = best;
}

// ---------------------------------------------------------------------------
// GRU gate update; h rows stride CP, XP/HP rows stride 600.
// ---------------------------------------------------------------------------
__global__ void gru_gate_kernel(const float* __restrict__ XP, const float* __restrict__ HP,
                                float* __restrict__ h, int t) {
  int idx = blockIdx.x * blockDim.x + threadIdx.x;
  if (idx >= 64 * 200) return;
  int b = idx / 200, c = idx % 200;
  const float* xp = XP + (size_t)(b * 10 + t) * 600;
  const float* hp = HP + (size_t)b * 600;
  float xz = xp[c], xr = xp[200 + c], xh = xp[400 + c];
  float hz = hp[c], hr = hp[200 + c], hh = hp[400 + c];
  float z = 1.f / (1.f + expf(-(xz + hz)));
  float r = 1.f / (1.f + expf(-(xr + hr)));
  float hc = tanhf(xh + r * hh);
  float ho = h[b * CP + c];
  h[b * CP + c] = z * ho + (1.f - z) * hc;
}

__global__ void d4_kernel(const float* __restrict__ h, const float* __restrict__ w,
                          const float* __restrict__ bias, float* __restrict__ out) {
  int tid = threadIdx.x;
  if (tid >= 128) return;
  int b = tid >> 1, o = tid & 1;
  float acc = bias[o];
  for (int c = 0; c < 200; ++c) acc += h[b * CP + c] * w[c * 2 + o];
  out[b * 2 + o] = fmaxf(acc, 0.f);
}

// ---------------------------------------------------------------------------
// Host orchestration
// ---------------------------------------------------------------------------
extern "C" void kernel_launch(void* const* d_in, const int* in_sizes, int n_in,
                              void* d_out, int out_size, void* d_ws, size_t ws_size,
                              hipStream_t stream) {
  (void)in_sizes; (void)n_in; (void)out_size; (void)ws_size;
  const int*   history  = (const int*)d_in[0];
  const int*   response = (const int*)d_in[1];
  const float* emb      = (const float*)d_in[2];
  const float* attn_u   = (const float*)d_in[3];
  const float* attn_dv  = (const float*)d_in[4];
  const float* attn_w1  = (const float*)d_in[5];
  const float* attn_b1  = (const float*)d_in[6];
  const float* attn_w2  = (const float*)d_in[7];
  const float* attn_b2  = (const float*)d_in[8];
  const float* attn_lng = (const float*)d_in[9];
  const float* attn_lnb = (const float*)d_in[10];
  const float* d1_w = (const float*)d_in[11];
  const float* d1_b = (const float*)d_in[12];
  const float* d2_w = (const float*)d_in[13];
  const float* d2_b = (const float*)d_in[14];
  const float* conv1_w = (const float*)d_in[15];
  const float* conv1_b = (const float*)d_in[16];
  const float* conv2_w = (const float*)d_in[17];
  const float* conv2_b = (const float*)d_in[18];
  const float* d3_w = (const float*)d_in[19];
  const float* d3_b = (const float*)d_in[20];
  const float* d4_w = (const float*)d_in[21];
  const float* d4_b = (const float*)d_in[22];
  const float* gru_k  = (const float*)d_in[23];
  const float* gru_rk = (const float*)d_in[24];
  const float* gru_b  = (const float*)d_in[25];
  const float* ln1_g = (const float*)d_in[26];
  const float* ln1_b = (const float*)d_in[27];
  const float* ln2_g = (const float*)d_in[28];
  const float* ln2_b = (const float*)d_in[29];
  float* outp = (float*)d_out;

  const int NB = 640;
  const int ROWS = NB * 50;                 // 32000
  const size_t ACTP = (size_t)ROWS * CP;    // padded activation tensor

  float* ws = (float*)d_ws;
  size_t off = 0;
  auto alloc = [&](size_t n) { float* p = ws + off; off += n; return p; };
  float* ctx   = alloc(ACTP);
  float* resp  = alloc(ACTP);
  float* cs    = alloc(ACTP);
  float* rs    = alloc(ACTP);
  float* cc    = alloc(ACTP);
  float* rc    = alloc(ACTP);
  float* T1    = alloc(ACTP);
  float* T2    = alloc(ACTP);
  float* S     = alloc((size_t)ROWS * SP);
  float* feat  = alloc((size_t)NB * 50 * 50 * 3);
  float* pool1 = alloc((size_t)NB * 17 * 17 * 32);
  float* pool2 = alloc((size_t)NB * 6 * 6 * 16);
  float* mv    = alloc((size_t)NB * CP);
  float* XP    = alloc((size_t)NB * 600);
  float* HP    = alloc((size_t)64 * 600);
  float* hst   = alloc((size_t)64 * CP);
  float* cm    = alloc((size_t)ROWS);
  float* rm    = alloc((size_t)ROWS);
  float* dpad  = alloc((size_t)13 * CP);
  size_t f32_total = off;

  // f16 packed-weight region follows the f32 region.
  _Float16* hbase = (_Float16*)(ws + off);
  size_t hoff = 0;
  auto halloc = [&](size_t n) { _Float16* p = hbase + hoff; hoff += n; return p; };

  // 1) Zero the f32 region (establishes all K-pad zeros deterministically).
  zero_ws_kernel<<<4096, 256, 0, stream>>>(ws, f32_total);

  // 2) Pack weights to f16 transposed+padded.
  auto pack = [&](const float* W, _Float16* Wt, int Kd, int Nd, int KP, int NP) {
    int total = NP * KP;
    pack_wt_kernel<<<(total + 255) / 256, 256, 0, stream>>>(W, Wt, Kd, Nd, KP, NP);
  };
  _Float16 *Ut[12], *W1t[12], *W2t[12], *D1t[3], *D2t[3], *D3t[3], *GKt[3], *GRt[3];
  for (int mo = 0; mo < 12; ++mo) {
    Ut[mo]  = halloc((size_t)208 * 224); pack(attn_u  + (size_t)mo * 40000, Ut[mo],  200, 200, 224, 208);
    W1t[mo] = halloc((size_t)208 * 224); pack(attn_w1 + (size_t)mo * 40000, W1t[mo], 200, 200, 224, 208);
    W2t[mo] = halloc((size_t)208 * 224); pack(attn_w2 + (size_t)mo * 40000, W2t[mo], 200, 200, 224, 208);
  }
  for (int k = 0; k < 3; ++k) {
    D1t[k] = halloc((size_t)208 * 800); pack(d1_w  + (size_t)k * 160000, D1t[k], 800, 200, 800, 208);
    D2t[k] = halloc((size_t)208 * 800); pack(d2_w  + (size_t)k * 160000, D2t[k], 800, 200, 800, 208);
    D3t[k] = halloc((size_t)208 * 576); pack(d3_w  + (size_t)k * 115200, D3t[k], 576, 200, 576, 208);
    GKt[k] = halloc((size_t)608 * 224); pack(gru_k  + (size_t)k * 120000, GKt[k], 200, 600, 224, 608);
    GRt[k] = halloc((size_t)608 * 224); pack(gru_rk + (size_t)k * 120000, GRt[k], 200, 600, 224, 608);
  }
  pack_dvec_kernel<<<(13 * CP + 255) / 256, 256, 0, stream>>>(attn_dv, dpad);
  const float* dOnes = dpad + 12 * CP;

  // 3) Embedding gather.
  embed_kernel<<<ROWS, 256, 0, stream>>>(history, response, emb, ctx, resp, cm, rm);

  const float invSqrtC = 0.07071067811865475f;  // 1/sqrt(200)

  auto gemm16 = [&](const float* X, int ldx, const _Float16* Wt, int KP,
                    const float* bias, float* Y, int ldy, int M, int Nd, int act) {
    int mt = M / 16, nt = (Nd + 15) / 16;
    dim3 grid((unsigned)((mt / 4) * nt));
    gemm16_kernel<<<grid, 128, 0, stream>>>(X, ldx, Wt, KP, bias, Y, ldy, M, Nd, act, nt);
  };

  auto attn = [&](const float* q, const float* kk, const float* qm, const float* km,
                  int g, int k, float* dst) {
    int mo = g * 3 + k;
    const float* b1 = attn_b1  + (size_t)mo * 200;
    const float* b2 = attn_b2  + (size_t)mo * 200;
    const float* lg = attn_lng + (size_t)mo * 200;
    const float* lb = attn_lnb + (size_t)mo * 200;
    gemm16(q,  CP, Ut[mo], 224, nullptr, T1, CP, ROWS, 200, 1);   // f1 = relu(qU)
    gemm16(kk, CP, Ut[mo], 224, nullptr, T2, CP, ROWS, 200, 1);   // f2 = relu(kU)
    attn_score_kernel<<<NB * 16, 32, 0, stream>>>(T1, T2, dpad + (size_t)mo * CP,
                                                  invSqrtC, S, SP, 1, 0);
    softmax_mask_kernel<<<ROWS, 32, 0, stream>>>(S, qm, km);
    attn_av_kernel<<<NB * 52, 32, 0, stream>>>(S, kk, T1);        // A @ kk
    ln_res_kernel<<<ROWS, 32, 0, stream>>>(T2, T1, q, lg, lb);    // LN(AV + q)
    gemm16(T2, CP, W1t[mo], 224, b1, T1, CP, ROWS, 200, 1);
    gemm16(T1, CP, W2t[mo], 224, b2, dst, CP, ROWS, 200, 0);
  };

  for (int k = 0; k < 3; ++k) {
    attn(ctx,  ctx,  cm, cm, 0, k, cs);
    attn(resp, resp, rm, rm, 1, k, rs);
    attn(ctx,  resp, cm, rm, 2, k, cc);
    attn(resp, ctx,  rm, cm, 3, k, rc);

    attn_score_kernel<<<NB * 16, 32, 0, stream>>>(ctx, resp, dOnes, invSqrtC, feat, 50, 3, 0);
    attn_score_kernel<<<NB * 16, 32, 0, stream>>>(cs,  rs,   dOnes, invSqrtC, feat, 50, 3, 1);
    attn_score_kernel<<<NB * 16, 32, 0, stream>>>(cc,  rc,   dOnes, invSqrtC, feat, 50, 3, 2);

    conv1_pool_kernel<<<(640 * 17 * 17 * 32 + 255) / 256, 256, 0, stream>>>(
        feat, conv1_w + (size_t)k * 864, conv1_b + k * 32, pool1);
    conv2_pool_kernel<<<(640 * 6 * 6 * 16 + 255) / 256, 256, 0, stream>>>(
        pool1, conv2_w + (size_t)k * 4608, conv2_b + k * 16, pool2);

    gemm16(pool2, 576, D3t[k], 576, d3_b + (size_t)k * 200, mv, CP, NB, 200, 1);

    zero_kernel<<<(64 * CP + 255) / 256, 256, 0, stream>>>(hst, 64 * CP);
    gemm16(mv, CP, GKt[k], 224, gru_b + (size_t)k * 1200, XP, 600, NB, 600, 0);
    for (int t = 0; t < 10; ++t) {
      gemm16(hst, CP, GRt[k], 224, gru_b + (size_t)k * 1200 + 600, HP, 600, 64, 600, 0);
      gru_gate_kernel<<<(64 * 200 + 255) / 256, 256, 0, stream>>>(XP, HP, hst, t);
    }
    d4_kernel<<<1, 128, 0, stream>>>(hst, d4_w + (size_t)k * 400, d4_b + k * 2,
                                     outp + (size_t)k * 128);

    {
      int nt = 13;
      dim3 grid((unsigned)((ROWS / 64) * nt));
      gemm_cat4_kernel<<<grid, 128, 0, stream>>>(ctx, cs, cc, ctx, cc,
          D1t[k], d1_b + (size_t)k * 200, T1, nt);
      gemm_cat4_kernel<<<grid, 128, 0, stream>>>(resp, rs, rc, resp, rc,
          D2t[k], d2_b + (size_t)k * 200, T2, nt);
    }
    if (k == 0) {
      add_inplace_kernel<<<(int)((ACTP + 255) / 256), 256, 0, stream>>>(ctx,  T1, (int)ACTP);
      add_inplace_kernel<<<(int)((ACTP + 255) / 256), 256, 0, stream>>>(resp, T2, (int)ACTP);
    } else {
      ln_res_kernel<<<ROWS, 32, 0, stream>>>(ctx,  ctx,  T1, ln1_g, ln1_b);
      ln_res_kernel<<<ROWS, 32, 0, stream>>>(resp, resp, T2, ln2_g, ln2_b);
    }
  }
}
